// TNTOODModel_77962246357281
// MI455X (gfx1250) — matrix operand order repaired and checked by reading, compiled, not run on hardware
//
#include <hip/hip_runtime.h>
#include <hip/hip_bf16.h>
#include <math.h>

typedef __bf16 v16bf __attribute__((ext_vector_type(16)));
typedef __bf16 v8bf  __attribute__((ext_vector_type(8)));
typedef __bf16 v4bf  __attribute__((ext_vector_type(4)));
typedef float  v8f   __attribute__((ext_vector_type(8)));

#define F_IN 384
#define HDIM 128
#define PDIM 128
#define RDIM 16

__device__ __forceinline__ unsigned enc_ord(float f) {
  unsigned u = __float_as_uint(f);
  return (u & 0x80000000u) ? ~u : (u | 0x80000000u);
}
__device__ __forceinline__ float dec_ord(unsigned e) {
  return (e & 0x80000000u) ? __uint_as_float(e & 0x7fffffffu) : __uint_as_float(~e);
}
__device__ __forceinline__ v16bf cat16(v8bf lo, v8bf hi) {
  return __builtin_shufflevector(lo, hi, 0, 1, 2, 3, 4, 5, 6, 7, 8, 9, 10, 11, 12, 13, 14, 15);
}
__device__ __forceinline__ v4bf f4_to_bf4(float4 f) {
  v4bf bv;
  bv[0] = (__bf16)f.x; bv[1] = (__bf16)f.y; bv[2] = (__bf16)f.z; bv[3] = (__bf16)f.w;
  return bv;
}

// ---------------- elementwise utilities ----------------
__global__ void k_f32_to_bf16(const float* __restrict__ in, __bf16* __restrict__ out, int n) {
  int i = blockIdx.x * 256 + threadIdx.x;
  if (i < n) out[i] = (__bf16)in[i];
}
__global__ void k_fill_f32(float* p, float v, int n) {
  int i = blockIdx.x * 256 + threadIdx.x;
  if (i < n) p[i] = v;
}
__global__ void k_fill_u32(unsigned* p, unsigned v, int n) {
  int i = blockIdx.x * 256 + threadIdx.x;
  if (i < n) p[i] = v;
}
__global__ void k_copy_f32(float* __restrict__ d, const float* __restrict__ s, int n) {
  int i = blockIdx.x * 256 + threadIdx.x;
  if (i < n) d[i] = s[i];
}

// ---------------- GCN degree / scatter (H = 128 compile-time) ----------------
__global__ void k_deg_count(const int* __restrict__ dst, float* __restrict__ deg, int E) {
  int e = blockIdx.x * 256 + threadIdx.x;
  if (e < E) atomicAdd(&deg[dst[e]], 1.0f);
}
__global__ void k_deg_rsqrt(float* __restrict__ deg, int n) {
  int i = blockIdx.x * 256 + threadIdx.x;
  if (i < n) deg[i] = rsqrtf(deg[i]);   // deg >= 1 (self loop)
}
// out[n,h] = dinv[n]^2 * xw[n,h] + b[h]
__global__ void k_gcn_self(const float* __restrict__ xw, const float* __restrict__ dinv,
                           const float* __restrict__ b, float* __restrict__ out, int total) {
  int i = blockIdx.x * 256 + threadIdx.x;
  if (i >= total) return;
  int r = i >> 7, h = i & (HDIM - 1);
  float di = dinv[r];
  out[i] = di * di * xw[i] + b[h];
}
// out[dst,h] += dinv[src]*dinv[dst]*xw[src,h]
__global__ void k_gcn_edge(const int* __restrict__ src, const int* __restrict__ dst,
                           const float* __restrict__ xw, const float* __restrict__ dinv,
                           float* __restrict__ out, int total) {
  int i = blockIdx.x * 256 + threadIdx.x;
  if (i >= total) return;
  int e = i >> 7, h = i & (HDIM - 1);
  int s = src[e], d = dst[e];
  atomicAdd(&out[d * HDIM + h], dinv[s] * dinv[d] * xw[s * HDIM + h]);
}

// ---------------- cross-fusion edge ops ----------------
__global__ __launch_bounds__(256) void k_edge_scores(const int* __restrict__ src, const int* __restrict__ dst,
                                                     const float* __restrict__ t, const float* __restrict__ g,
                                                     float* __restrict__ scores, int E) {
  int e = blockIdx.x * 8 + (threadIdx.x >> 5);
  int lane = threadIdx.x & 31;
  if (e >= E) return;
  const float* ts = t + (long long)src[e] * F_IN;
  const float* gd = g + (long long)dst[e] * F_IN;
  float acc = 0.0f;
#pragma unroll
  for (int i = 0; i < F_IN / 32; ++i) acc += ts[lane + i * 32] * gd[lane + i * 32];
#pragma unroll
  for (int m = 16; m; m >>= 1) acc += __shfl_xor(acc, m, 32);
  if (lane == 0) scores[e] = acc * 0.05103103630798288f; // 1/sqrt(384)
}
__global__ void k_seg_max(const int* __restrict__ src, const float* __restrict__ scores,
                          unsigned* __restrict__ mu, int E) {
  int e = blockIdx.x * 256 + threadIdx.x;
  if (e < E) atomicMax(&mu[src[e]], enc_ord(scores[e]));
}
__global__ void k_m_decode(const unsigned* __restrict__ mu, float* __restrict__ mf, int n) {
  int i = blockIdx.x * 256 + threadIdx.x;
  if (i < n) mf[i] = (mu[i] == 0u) ? -INFINITY : dec_ord(mu[i]);
}
__global__ void k_edge_exp(const int* __restrict__ src, const float* __restrict__ scores,
                           const float* __restrict__ mf, float* __restrict__ ebuf,
                           float* __restrict__ denom, int E) {
  int e = blockIdx.x * 256 + threadIdx.x;
  if (e >= E) return;
  int s = src[e];
  float v = __expf(scores[e] - mf[s]);
  ebuf[e] = v;
  atomicAdd(&denom[s], v);
}
__global__ void k_fusion_scatter(const int* __restrict__ src, const int* __restrict__ dst,
                                 const float* __restrict__ ebuf, const float* __restrict__ denom,
                                 const float* __restrict__ v, float* __restrict__ xf, int total) {
  int i = blockIdx.x * 256 + threadIdx.x;
  if (i >= total) return;
  int e = i / F_IN, f = i - e * F_IN;
  int s = src[e], d = dst[e];
  float attn = ebuf[e] / denom[s];
  atomicAdd(&xf[s * F_IN + f], attn * v[(long long)d * F_IN + f]);
}

// ---------------- WMMA GEMM (pipelined, double-buffered) ----------------
// C[MxNc] = A[MxK](f32->bf16) @ B[KxNc](bf16) + bias, opt relu.
// 256 threads (8 waves) compute 128x64; wave w: column tile (w&3), 4 row tiles,
// one shared B fragment -> 4 v_wmma per k-step. Ping-pong LDS: global loads for
// step k+1 are issued before the WMMAs of step k; only the LDS stores wait.
// Tail A rows are address-clamped (not zeroed): garbage rows feed only
// unstored C rows, so no divergent staging branches.
__global__ __launch_bounds__(256) void k_wmma_gemm(const float* __restrict__ A,
                                                   const __bf16* __restrict__ B,
                                                   const float* __restrict__ bias,
                                                   float* __restrict__ C,
                                                   int M, int K, int Ncols, int relu) {
  __shared__ __bf16 As[2][128][40];   // [buf][row][k] rows 80B: aligned, conflict-free b128
  __shared__ __bf16 BsT[2][64][40];   // [buf][col][k] transposed
  const int tid = threadIdx.x;
  const int wave = tid >> 5;
  const int lane = tid & 31;
  const int lidx = lane & 15;
  const int lhalf = lane >> 4;
  const int rowBase = blockIdx.y * 128;
  const int colBase = blockIdx.x * 64;
  const int tc = wave & 3;
  const int trBase = (wave >> 2) * 4;
  const int kbA = lhalf * 8;
  const int kbB = lhalf * 8;
  const int nk = K >> 5;

  // per-thread staging coordinates
  const int ra0 = tid >> 3, ca0 = (tid & 7) * 4;            // A chunk 0 of 4 (stride 32 rows)
  const int rb0 = tid >> 4, cb0 = (tid & 15) * 4;           // B chunk 0 of 2 (stride 16 rows)

  float4 fa[4];
  v4bf fb[2];

  auto load_tiles = [&](int k0) {
#pragma unroll
    for (int t = 0; t < 4; ++t) {
      int gr = rowBase + ra0 + t * 32;
      if (gr >= M) gr = M - 1;                              // clamp, no branch
      fa[t] = *(const float4*)&A[(long long)gr * K + k0 + ca0];
    }
#pragma unroll
    for (int t = 0; t < 2; ++t)
      fb[t] = *(const v4bf*)&B[(long long)(k0 + rb0 + t * 16) * Ncols + colBase + cb0];
  };
  auto stage_tiles = [&](int buf) {
#pragma unroll
    for (int t = 0; t < 4; ++t)
      *(v4bf*)&As[buf][ra0 + t * 32][ca0] = f4_to_bf4(fa[t]);
#pragma unroll
    for (int t = 0; t < 2; ++t) {
      int r = rb0 + t * 16;
#pragma unroll
      for (int q = 0; q < 4; ++q) BsT[buf][cb0 + q][r] = fb[t][q];
    }
  };

  v8f acc[4] = {};

  load_tiles(0);
  stage_tiles(0);

  for (int k = 0; k < nk; ++k) {
    __syncthreads();                                        // buf (k&1) ready for all waves
    if (k + 1 < nk) load_tiles((k + 1) << 5);               // issue globals, no wait yet
    const int buf = k & 1;
    const int n = tc * 16 + lidx;
    v16bf bfrag = cat16(*(const v8bf*)&BsT[buf][n][kbB], *(const v8bf*)&BsT[buf][n][16 + kbB]);
    v16bf af[4];
#pragma unroll
    for (int i = 0; i < 4; ++i) {
      const int m = (trBase + i) * 16 + lidx;
      af[i] = cat16(*(const v8bf*)&As[buf][m][kbA], *(const v8bf*)&As[buf][m][16 + kbA]);
    }
#pragma unroll
    for (int i = 0; i < 4; ++i)
      acc[i] = __builtin_amdgcn_wmma_f32_16x16x32_bf16(false, af[i], false, bfrag,
                                                       (short)0, acc[i], false, false);
    if (k + 1 < nk) stage_tiles((k + 1) & 1);               // waits loadcnt, fills other buf
  }

  const int col = colBase + tc * 16 + lidx;
  const float bvv = bias ? bias[col] : 0.0f;
#pragma unroll
  for (int i = 0; i < 4; ++i) {
#pragma unroll
    for (int v = 0; v < 8; ++v) {
      int row = rowBase + (trBase + i) * 16 + v + lhalf * 8;
      if (row < M) {
        float o = acc[i][v] + bvv;
        if (relu) o = fmaxf(o, 0.0f);
        C[(long long)row * Ncols + col] = o;
      }
    }
  }
}

// ---------------- fused low-rank contraction ----------------
// out[n,u] = sum_{j<G} ((A@W)[n, u*G+j] + bias[u*G+j]) * mul[n,j]
// temp pass: A=h_r (Nx192), W=Wr2 (192x6144), U=16,  G=384, mul=x    -> temp (Nx16)
// p_t  pass: A=h_l (Nx192), W=Wl2 (192x2048), U=128, G=16,  mul=temp -> p_t  (Nx128)
// The (A@W) 16x16 WMMA tiles are consumed in-register; 'right'/'left' never hit HBM.
__global__ __launch_bounds__(128) void k_wmma_lowrank(const float* __restrict__ A,
                                                      const __bf16* __restrict__ W,
                                                      const float* __restrict__ bias,
                                                      const float* __restrict__ mul,
                                                      float* __restrict__ out,
                                                      int Nn, int K, int U, int G) {
  __shared__ __bf16 As[16][40];
  __shared__ __bf16 WsT[384][40];   // [col-in-group][k] transposed; G<=384
  __shared__ float accRow[16];
  const int tid = threadIdx.x;
  const int wave = tid >> 5;
  const int lane = tid & 31;
  const int lidx = lane & 15;
  const int lhalf = lane >> 4;
  const int tileBase = blockIdx.x * 16;
  const int u = blockIdx.y;
  const int WN = U * G;
  const int nct = G >> 4;
  const int kbA = lhalf * 8;
  const int kbB = lhalf * 8;

  v8f acc[6] = {};

  for (int k0 = 0; k0 < K; k0 += 32) {
    // stage A: 16x32 f32 -> bf16 (one float4 per thread; clamped rows, no branch)
    {
      int r = tid >> 3, c4 = (tid & 7) * 4;
      int gr = tileBase + r;
      if (gr >= Nn) gr = Nn - 1;
      float4 f = *(const float4*)&A[(long long)gr * K + k0 + c4];
      *(v4bf*)&As[r][c4] = f4_to_bf4(f);
    }
    // stage W transposed: 32 x G bf16
    for (int idx = tid; idx < 8 * G; idx += 128) {   // 32*G/4 v4bf loads
      int r = idx / (G >> 2), c = (idx % (G >> 2)) * 4;
      v4bf bv = *(const v4bf*)&W[(long long)(k0 + r) * WN + u * G + c];
      WsT[c + 0][r] = bv[0];
      WsT[c + 1][r] = bv[1];
      WsT[c + 2][r] = bv[2];
      WsT[c + 3][r] = bv[3];
    }
    if (k0 + 32 < K && tid < 32)
      __builtin_prefetch(&W[(long long)(k0 + 32 + tid) * WN + u * G], 0, 1);
    __syncthreads();

    v16bf afrag = cat16(*(const v8bf*)&As[lidx][kbA], *(const v8bf*)&As[lidx][16 + kbA]);
#pragma unroll
    for (int i = 0; i < 6; ++i) {
      int ct = wave + i * 4;
      if (ct < nct) {
        const int n = ct * 16 + lidx;
        v16bf bfrag = cat16(*(const v8bf*)&WsT[n][kbB], *(const v8bf*)&WsT[n][16 + kbB]);
        acc[i] = __builtin_amdgcn_wmma_f32_16x16x32_bf16(false, afrag, false, bfrag,
                                                         (short)0, acc[i], false, false);
      }
    }
    __syncthreads();
  }

  if (tid < 16) accRow[tid] = 0.0f;
  __syncthreads();

#pragma unroll
  for (int i = 0; i < 6; ++i) {
    int ct = wave + i * 4;
    if (ct < nct) {
      int colg = ct * 16 + lidx;
      float bv = bias[u * G + colg];
#pragma unroll
      for (int v = 0; v < 8; ++v) {
        int rl = v + lhalf * 8;
        int node = tileBase + rl;
        float mv = (node < Nn) ? mul[(long long)node * G + colg] : 0.0f;
        atomicAdd(&accRow[rl], (acc[i][v] + bv) * mv);   // ds_add_f32
      }
    }
  }
  __syncthreads();
  if (tid < 16) {
    int node = tileBase + tid;
    if (node < Nn) out[(long long)node * U + u] = accRow[tid];
  }
}

// ---------------- finalize: normalize p_t, p_g; alignment ----------------
__global__ __launch_bounds__(256) void k_finalize(const float* __restrict__ pt, const float* __restrict__ pg,
                                                  float* __restrict__ oalign, float* __restrict__ opt,
                                                  float* __restrict__ opg, int Nn) {
  int node = blockIdx.x * 8 + (threadIdx.x >> 5);
  int lane = threadIdx.x & 31;
  if (node >= Nn) return;
  float a[4], b[4], sa = 0.0f, sb = 0.0f;
#pragma unroll
  for (int i = 0; i < 4; ++i) {
    int idx = lane + i * 32;
    a[i] = pt[(long long)node * PDIM + idx];
    b[i] = pg[(long long)node * PDIM + idx];
    sa += a[i] * a[i];
    sb += b[i] * b[i];
  }
#pragma unroll
  for (int m = 16; m; m >>= 1) { sa += __shfl_xor(sa, m, 32); sb += __shfl_xor(sb, m, 32); }
  float ia = 1.0f / fmaxf(sqrtf(sa), 1e-12f);
  float ib = 1.0f / fmaxf(sqrtf(sb), 1e-12f);
  float dotv = 0.0f;
#pragma unroll
  for (int i = 0; i < 4; ++i) {
    int idx = lane + i * 32;
    float na = a[i] * ia, nb = b[i] * ib;
    opt[(long long)node * PDIM + idx] = na;
    opg[(long long)node * PDIM + idx] = nb;
    dotv += na * nb;
  }
#pragma unroll
  for (int m = 16; m; m >>= 1) dotv += __shfl_xor(dotv, m, 32);
  if (lane == 0) oalign[node] = dotv;
}

// ---------------- host orchestration ----------------
static inline unsigned cdiv(long long a, long long b) { return (unsigned)((a + b - 1) / b); }

extern "C" void kernel_launch(void* const* d_in, const int* in_sizes, int n_in,
                              void* d_out, int out_size, void* d_ws, size_t ws_size,
                              hipStream_t stream) {
  const float* x      = (const float*)d_in[0];
  const int*   ei     = (const int*)d_in[1];
  const float* W_gcn1 = (const float*)d_in[2];  const float* b_gcn1 = (const float*)d_in[3];
  const float* W_ti   = (const float*)d_in[4];  const float* b_ti   = (const float*)d_in[5];
  const float* W_gi   = (const float*)d_in[6];  const float* b_gi   = (const float*)d_in[7];
  const float* W_v    = (const float*)d_in[8];  const float* b_v    = (const float*)d_in[9];
  const float* Wl1    = (const float*)d_in[10]; const float* bl1    = (const float*)d_in[11];
  const float* Wl2    = (const float*)d_in[12]; const float* bl2    = (const float*)d_in[13];
  const float* Wr1    = (const float*)d_in[14]; const float* br1    = (const float*)d_in[15];
  const float* Wr2    = (const float*)d_in[16]; const float* br2    = (const float*)d_in[17];
  const float* W_gcn2 = (const float*)d_in[18]; const float* b_gcn2 = (const float*)d_in[19];
  const float* W_gp   = (const float*)d_in[20]; const float* b_gp   = (const float*)d_in[21];

  const int N = in_sizes[0] / F_IN;
  const int E = in_sizes[1] / 2;
  const int* src = ei;       // row
  const int* dst = ei + E;   // col

  // output layout: alignment[N], p_t[N*128], p_g[N*128], g2[N*128]
  float* out      = (float*)d_out;
  float* o_align  = out;
  float* o_pt     = out + N;
  float* o_pg     = out + N + (long long)N * PDIM;
  float* o_g2     = out + N + 2LL * N * PDIM;

  char* base = (char*)d_ws;
  size_t off = 0;
  auto carve = [&](size_t bytes) -> void* {
    size_t o = (off + 255) & ~(size_t)255;
    off = o + bytes;
    return (void*)(base + o);
  };
  float*    xw     = (float*)carve((size_t)N * HDIM * 4);   // reused as p_t raw
  float*    dinv   = (float*)carve((size_t)N * 4);
  float*    g_emb  = (float*)carve((size_t)N * HDIM * 4);   // reused as p2w
  float*    t_info = (float*)carve((size_t)N * F_IN * 4);   // reused as h_l
  float*    g_info = (float*)carve((size_t)N * F_IN * 4);   // reused as h_r
  float*    vbuf   = (float*)carve((size_t)N * F_IN * 4);   // reused as p_g raw
  float*    xf     = (float*)carve((size_t)N * F_IN * 4);
  float*    scores = (float*)carve((size_t)E * 4);
  float*    ebuf   = (float*)carve((size_t)E * 4);
  unsigned* mu     = (unsigned*)carve((size_t)N * 4);
  float*    mf     = (float*)carve((size_t)N * 4);
  float*    denom  = (float*)carve((size_t)N * 4);
  float*    temp   = (float*)carve((size_t)N * RDIM * 4);
  __bf16* wb_gcn1 = (__bf16*)carve((size_t)F_IN * HDIM * 2);
  __bf16* wb_ti   = (__bf16*)carve((size_t)F_IN * F_IN * 2);
  __bf16* wb_gi   = (__bf16*)carve((size_t)HDIM * F_IN * 2);
  __bf16* wb_v    = (__bf16*)carve((size_t)F_IN * F_IN * 2);
  __bf16* wb_l1   = (__bf16*)carve((size_t)F_IN * 192 * 2);
  __bf16* wb_l2   = (__bf16*)carve((size_t)192 * PDIM * RDIM * 2);
  __bf16* wb_r1   = (__bf16*)carve((size_t)F_IN * 192 * 2);
  __bf16* wb_r2   = (__bf16*)carve((size_t)192 * RDIM * F_IN * 2);
  __bf16* wb_gcn2 = (__bf16*)carve((size_t)PDIM * HDIM * 2);
  __bf16* wb_gp   = (__bf16*)carve((size_t)PDIM * PDIM * 2);
  float* h_l  = t_info;   // N x 192
  float* h_r  = g_info;   // N x 192
  float* p_t  = xw;       // N x 128 (raw)
  float* p2w  = g_emb;    // N x 128
  float* pg_r = vbuf;     // N x 128 (raw p_g)

  // 0) weights -> bf16
  k_f32_to_bf16<<<cdiv(F_IN * HDIM, 256), 256, 0, stream>>>(W_gcn1, wb_gcn1, F_IN * HDIM);
  k_f32_to_bf16<<<cdiv(F_IN * F_IN, 256), 256, 0, stream>>>(W_ti, wb_ti, F_IN * F_IN);
  k_f32_to_bf16<<<cdiv(HDIM * F_IN, 256), 256, 0, stream>>>(W_gi, wb_gi, HDIM * F_IN);
  k_f32_to_bf16<<<cdiv(F_IN * F_IN, 256), 256, 0, stream>>>(W_v, wb_v, F_IN * F_IN);
  k_f32_to_bf16<<<cdiv(F_IN * 192, 256), 256, 0, stream>>>(Wl1, wb_l1, F_IN * 192);
  k_f32_to_bf16<<<cdiv(192 * PDIM * RDIM, 256), 256, 0, stream>>>(Wl2, wb_l2, 192 * PDIM * RDIM);
  k_f32_to_bf16<<<cdiv(F_IN * 192, 256), 256, 0, stream>>>(Wr1, wb_r1, F_IN * 192);
  k_f32_to_bf16<<<cdiv(192 * RDIM * F_IN, 256), 256, 0, stream>>>(Wr2, wb_r2, 192 * RDIM * F_IN);
  k_f32_to_bf16<<<cdiv(PDIM * HDIM, 256), 256, 0, stream>>>(W_gcn2, wb_gcn2, PDIM * HDIM);
  k_f32_to_bf16<<<cdiv(PDIM * PDIM, 256), 256, 0, stream>>>(W_gp, wb_gp, PDIM * PDIM);

  // 1) xw = x @ W_gcn1 (bias added in scatter)
  k_wmma_gemm<<<dim3(HDIM / 64, cdiv(N, 128)), 256, 0, stream>>>(x, wb_gcn1, nullptr, xw, N, F_IN, HDIM, 0);
  // 2) degrees (self-loop => init 1), dinv = rsqrt(deg)
  k_fill_f32<<<cdiv(N, 256), 256, 0, stream>>>(dinv, 1.0f, N);
  k_deg_count<<<cdiv(E, 256), 256, 0, stream>>>(dst, dinv, E);
  k_deg_rsqrt<<<cdiv(N, 256), 256, 0, stream>>>(dinv, N);
  // 3) g_emb = scatter-norm(xw) + b_gcn1
  k_gcn_self<<<cdiv((long long)N * HDIM, 256), 256, 0, stream>>>(xw, dinv, b_gcn1, g_emb, N * HDIM);
  k_gcn_edge<<<cdiv((long long)E * HDIM, 256), 256, 0, stream>>>(src, dst, xw, dinv, g_emb, E * HDIM);
  // 4) t_info, g_info, v
  k_wmma_gemm<<<dim3(F_IN / 64, cdiv(N, 128)), 256, 0, stream>>>(x, wb_ti, b_ti, t_info, N, F_IN, F_IN, 0);
  k_wmma_gemm<<<dim3(F_IN / 64, cdiv(N, 128)), 256, 0, stream>>>(g_emb, wb_gi, b_gi, g_info, N, HDIM, F_IN, 0);
  k_wmma_gemm<<<dim3(F_IN / 64, cdiv(N, 128)), 256, 0, stream>>>(x, wb_v, b_v, vbuf, N, F_IN, F_IN, 0);
  // 5) edge scores + scatter softmax over src
  k_edge_scores<<<cdiv(E, 8), 256, 0, stream>>>(src, dst, t_info, g_info, scores, E);
  k_fill_u32<<<cdiv(N, 256), 256, 0, stream>>>(mu, 0u, N);
  k_seg_max<<<cdiv(E, 256), 256, 0, stream>>>(src, scores, mu, E);
  k_m_decode<<<cdiv(N, 256), 256, 0, stream>>>(mu, mf, N);
  k_fill_f32<<<cdiv(N, 256), 256, 0, stream>>>(denom, 0.0f, N);
  k_edge_exp<<<cdiv(E, 256), 256, 0, stream>>>(src, scores, mf, ebuf, denom, E);
  // 6) x_fused = x + scatter(attn * v[dst]) at src
  k_copy_f32<<<cdiv((long long)N * F_IN, 256), 256, 0, stream>>>(xf, x, N * F_IN);
  k_fusion_scatter<<<cdiv((long long)E * F_IN, 256), 256, 0, stream>>>(src, dst, ebuf, denom, vbuf, xf, E * F_IN);
  // 7) generators' hidden layers
  k_wmma_gemm<<<dim3(192 / 64, cdiv(N, 128)), 256, 0, stream>>>(xf, wb_l1, bl1, h_l, N, F_IN, 192, 1);
  k_wmma_gemm<<<dim3(192 / 64, cdiv(N, 128)), 256, 0, stream>>>(xf, wb_r1, br1, h_r, N, F_IN, 192, 1);
  // 8) fused low-rank projection
  k_wmma_lowrank<<<dim3(cdiv(N, 16), RDIM), 128, 0, stream>>>(h_r, wb_r2, br2, x, temp, N, 192, RDIM, F_IN);
  k_wmma_lowrank<<<dim3(cdiv(N, 16), PDIM), 128, 0, stream>>>(h_l, wb_l2, bl2, temp, p_t, N, 192, PDIM, RDIM);
  // 9) gcn2: g2 written straight into d_out
  k_wmma_gemm<<<dim3(HDIM / 64, cdiv(N, 128)), 256, 0, stream>>>(p_t, wb_gcn2, nullptr, p2w, N, PDIM, HDIM, 0);
  k_gcn_self<<<cdiv((long long)N * HDIM, 256), 256, 0, stream>>>(p2w, dinv, b_gcn2, o_g2, N * HDIM);
  k_gcn_edge<<<cdiv((long long)E * HDIM, 256), 256, 0, stream>>>(src, dst, p2w, dinv, o_g2, E * HDIM);
  // 10) p_g raw = g2 @ W_gp + b_gp
  k_wmma_gemm<<<dim3(PDIM / 64, cdiv(N, 128)), 256, 0, stream>>>(o_g2, wb_gp, b_gp, pg_r, N, HDIM, PDIM, 0);
  // 11) normalize + alignment
  k_finalize<<<cdiv(N, 8), 256, 0, stream>>>(p_t, pg_r, o_align, o_pt, o_pg, N);

  (void)n_in; (void)out_size; (void)ws_size;
}
